// RTNTM_48988396978470
// MI455X (gfx1250) — compile-verified, matrix operand order checked by reading, not run on hardware
//
#include <hip/hip_runtime.h>
#include <cstdint>
#include <cstddef>

// ---------------- constants (match reference) ----------------
enum : int { T_ = 8, B_ = 32, V_ = 50257, D_ = 256, N_ = 4096, H_ = 4, L_ = 2, K_ = 3, WIN_ = 16 };
// read param stride = D+3+K = 262 ; write param stride = 3D+3+K = 774

#define DEV __device__ __forceinline__

typedef __attribute__((ext_vector_type(16))) __bf16 v16bf;
typedef __attribute__((ext_vector_type(8)))  __bf16 v8bf;
typedef __attribute__((ext_vector_type(8)))  float  v8f;

DEV __bf16 f2bf(float f) {
  unsigned u = __builtin_bit_cast(unsigned, f);
  u += 0x7fffu + ((u >> 16) & 1u);                 // round-to-nearest-even
  unsigned short h = (unsigned short)(u >> 16);
  return __builtin_bit_cast(__bf16, h);
}
DEV float sigf(float x)      { return 1.f / (1.f + __expf(-x)); }
DEV float softplusf(float x) { return (x > 20.f) ? x : log1pf(__expf(x)); }

// --- CDNA5 async global->LDS copy (ASYNCcnt path, CDNA5 ISA), via inline asm ---
DEV void async_ld16(unsigned lds_addr, const float* g) {
  asm volatile("global_load_async_to_lds_b128 %0, %1, off"
               :: "v"(lds_addr), "v"(g) : "memory");
}
DEV void async_wait0() {
  asm volatile("s_wait_asynccnt 0" ::: "memory");
}

// ---------------- utility kernels ----------------
__global__ void k_zero(float* p, size_t n) {
  size_t i = (size_t)blockIdx.x * blockDim.x + threadIdx.x;
  size_t st = (size_t)gridDim.x * blockDim.x;
  for (; i < n; i += st) p[i] = 0.f;
}

__global__ void k_initw(float* p, size_t n) {   // one-hot at n%N==0
  size_t i = (size_t)blockIdx.x * blockDim.x + threadIdx.x;
  size_t st = (size_t)gridDim.x * blockDim.x;
  for (; i < n; i += st) p[i] = ((i & (size_t)(N_ - 1)) == 0) ? 1.f : 0.f;
}

__global__ void k_cvt_bf16(const float* __restrict__ s, unsigned short* __restrict__ d, size_t n) {
  size_t i = (size_t)blockIdx.x * blockDim.x + threadIdx.x;
  size_t st = (size_t)gridDim.x * blockDim.x;
  for (; i < n; i += st) {
    unsigned u = __builtin_bit_cast(unsigned, s[i]);
    u += 0x7fffu + ((u >> 16) & 1u);
    d[i] = (unsigned short)(u >> 16);
  }
}

__global__ void k_add(float* __restrict__ y, const float* __restrict__ x, size_t n) {
  size_t i = (size_t)blockIdx.x * blockDim.x + threadIdx.x;
  if (i < n) y[i] += x[i];
}
__global__ void k_add2(float* __restrict__ o, const float* __restrict__ a,
                       const float* __restrict__ b, size_t n) {
  size_t i = (size_t)blockIdx.x * blockDim.x + threadIdx.x;
  if (i < n) o[i] = a[i] + b[i];
}
__global__ void k_relu(float* __restrict__ x, size_t n) {
  size_t i = (size_t)blockIdx.x * blockDim.x + threadIdx.x;
  if (i < n) x[i] = fmaxf(x[i], 0.f);
}

// ---------------- WMMA bf16 GEMM: C[M,Nn] = A[M,K](f32) @ W[Nn,K](bf16)^T + bias ------
// block = 256 threads = 8 wave32 waves. Each wave owns a 16(M) x 64(N) output slab:
// one A fragment feeds FOUR v_wmma_f32_16x16x32_bf16 per K-step. Block tile: 32 x 256.
// K is a template constant -> fully unrolled K-loop (32..128 static WMMAs).
// Preconditions (all call sites): M % 32 == 0, bias != nullptr, K % 32 == 0.

#define GEMM_BTILE(WPTR, ACC)                                              \
  {                                                                        \
    uint4 b0 = *(const uint4*)(WPTR + k0);                                 \
    uint4 b1 = *(const uint4*)(WPTR + k1);                                 \
    v8bf w0 = __builtin_bit_cast(v8bf, b0);                                \
    v8bf w1 = __builtin_bit_cast(v8bf, b1);                                \
    v16bf wf;                                                              \
    wf[0]=w0[0];  wf[1]=w0[1];  wf[2]=w0[2];  wf[3]=w0[3];                 \
    wf[4]=w0[4];  wf[5]=w0[5];  wf[6]=w0[6];  wf[7]=w0[7];                 \
    wf[8]=w1[0];  wf[9]=w1[1];  wf[10]=w1[2]; wf[11]=w1[3];                \
    wf[12]=w1[4]; wf[13]=w1[5]; wf[14]=w1[6]; wf[15]=w1[7];                \
    ACC = __builtin_amdgcn_wmma_f32_16x16x32_bf16(false, af, false, wf,    \
                                                  (short)0, ACC, false, false); \
  }

#define GEMM_STORE(JT, ACC)                                                \
  {                                                                        \
    const int cn = nbase + (JT) * 16 + m16;                                \
    if (cn < Nn) {                                                         \
      const float bv = bias[cn];                                           \
      float* Cp = C + mb * Nn + cn;                                        \
      Cp[0*Nn] = ACC[0] + bv; Cp[1*Nn] = ACC[1] + bv;                      \
      Cp[2*Nn] = ACC[2] + bv; Cp[3*Nn] = ACC[3] + bv;                      \
      Cp[4*Nn] = ACC[4] + bv; Cp[5*Nn] = ACC[5] + bv;                      \
      Cp[6*Nn] = ACC[6] + bv; Cp[7*Nn] = ACC[7] + bv;                      \
    }                                                                      \
  }

template<int K>
__global__ __launch_bounds__(256) void k_gemm_bf16(
    const float* __restrict__ A, const unsigned short* __restrict__ W,
    const float* __restrict__ bias, float* __restrict__ C,
    int M, int Nn)
{
  const int lane = threadIdx.x & 31;
  const int wave = threadIdx.x >> 5;
  const int hf   = lane >> 4;                 // lane half selects K sub-pattern (ISA 7.12.2)
  const int m16  = lane & 15;

  const int tileM = blockIdx.x * 32 + (wave >> 2) * 16;          // tileM+15 < M guaranteed
  const int nbase = blockIdx.y * 256 + (wave & 3) * 64;          // wave's 64-wide N slab

  const float* Arow = A + (size_t)(tileM + m16) * K;
  int wn0 = nbase +  0 + m16; if (wn0 >= Nn) wn0 = Nn - 1;       // clamped loads
  int wn1 = nbase + 16 + m16; if (wn1 >= Nn) wn1 = Nn - 1;
  int wn2 = nbase + 32 + m16; if (wn2 >= Nn) wn2 = Nn - 1;
  int wn3 = nbase + 48 + m16; if (wn3 >= Nn) wn3 = Nn - 1;
  const unsigned short* Wr0 = W + (size_t)wn0 * K;
  const unsigned short* Wr1 = W + (size_t)wn1 * K;
  const unsigned short* Wr2 = W + (size_t)wn2 * K;
  const unsigned short* Wr3 = W + (size_t)wn3 * K;

  v8f acc0 = {}, acc1 = {}, acc2 = {}, acc3 = {};

  #pragma unroll
  for (int kb = 0; kb < K; kb += 32) {
    const int k0 = kb + (hf ? 8 : 0);
    const int k1 = k0 + 16;
    // A fragment: 16 bf16 per lane (row of A, K-striped per ISA layout), reused 4x
    float4 a0 = *(const float4*)(Arow + k0);
    float4 a1 = *(const float4*)(Arow + k0 + 4);
    float4 a2 = *(const float4*)(Arow + k1);
    float4 a3 = *(const float4*)(Arow + k1 + 4);
    v16bf af;
    af[0]=f2bf(a0.x);  af[1]=f2bf(a0.y);  af[2]=f2bf(a0.z);  af[3]=f2bf(a0.w);
    af[4]=f2bf(a1.x);  af[5]=f2bf(a1.y);  af[6]=f2bf(a1.z);  af[7]=f2bf(a1.w);
    af[8]=f2bf(a2.x);  af[9]=f2bf(a2.y);  af[10]=f2bf(a2.z); af[11]=f2bf(a2.w);
    af[12]=f2bf(a3.x); af[13]=f2bf(a3.y); af[14]=f2bf(a3.z); af[15]=f2bf(a3.w);

    GEMM_BTILE(Wr0, acc0)
    GEMM_BTILE(Wr1, acc1)
    GEMM_BTILE(Wr2, acc2)
    GEMM_BTILE(Wr3, acc3)
  }

  const int mb = tileM + (hf ? 8 : 0);        // 32-bit index math throughout
  GEMM_STORE(0, acc0)
  GEMM_STORE(1, acc1)
  GEMM_STORE(2, acc2)
  GEMM_STORE(3, acc3)
}

// ---------------- r = read_w . memory  (B x D; streams 134MB) ----------------
// Rows staged global->LDS with CDNA5 async copies: lanes 0..63 (waves 0-1) issue
// global_load_async_to_lds_b128 for row n+1 while all 256 threads FMA row n from LDS.
__global__ __launch_bounds__(256) void k_read_r(const float* __restrict__ mem,
                                                const float* __restrict__ rw,
                                                float* __restrict__ r)
{
  __shared__ float buf[2][D_];                 // 2KB double buffer
  const int b = blockIdx.x, tid = threadIdx.x;
  const float* mbase = mem + (size_t)b * N_ * D_;
  const float* wb = rw + (size_t)b * N_;

  const bool loader = (tid < 64);              // wave-uniform: waves 0 and 1
  if (loader) {                                // prime row 0
    async_ld16((unsigned)(size_t)&buf[0][tid * 4], mbase + tid * 4);
    async_wait0();
  }
  __syncthreads();

  float acc = 0.f;
  for (int n = 0; n < N_; ++n) {
    const int cur = n & 1;
    const bool more = (n + 1) < N_;
    if (loader && more)                        // kick off next row
      async_ld16((unsigned)(size_t)&buf[cur ^ 1][tid * 4],
                 mbase + (size_t)(n + 1) * D_ + tid * 4);
    acc = fmaf(wb[n], buf[cur][tid], acc);
    if (loader && more) async_wait0();         // ASYNCcnt drain before handoff
    __syncthreads();
  }
  r[b * D_ + tid] = acc;
}

// ---------------- x_in = gam*emb + bet + rg*r -> hist slot t+1 ----------------
__global__ void k_xin(const int* __restrict__ tok, const float* __restrict__ embW,
                      const float* __restrict__ fp, const float* __restrict__ rgraw,
                      const float* __restrict__ r, float* __restrict__ hist, int t)
{
  const int b = blockIdx.x, d = threadIdx.x;
  const int token = tok[t * B_ + b];
  const float e   = embW[(size_t)token * D_ + d];
  const float gam = 1.f + tanhf(fp[b * 2 * D_ + d]);
  const float bet = fp[b * 2 * D_ + D_ + d];
  const float rg  = sigf(rgraw[b * D_ + d]);
  hist[((size_t)(t + 1) * B_ + b) * D_ + d] = gam * e + bet + rg * r[b * D_ + d];
}

// ---------------- h = ctx + positional encoding ----------------
__global__ void k_build_h(const float* __restrict__ hist, float* __restrict__ h)
{
  const int row = blockIdx.x;             // s*B + b
  const int s = row >> 5;
  const int d = threadIdx.x;
  const int i = d >> 1;
  const float div = __expf(-(float)(2 * i) * (9.210340372f / (float)D_));
  const float ang = (float)s * div;
  const float pe = (d & 1) ? __cosf(ang) : __sinf(ang);
  h[(size_t)row * D_ + d] = hist[(size_t)row * D_ + d] + pe;
}

// ---------------- LayerNorm (one wave32 per row of 256) ----------------
__global__ void k_ln(const float* __restrict__ X, float* __restrict__ Y,
                     const float* __restrict__ g, const float* __restrict__ be, int rows)
{
  const int row  = blockIdx.x * 8 + (threadIdx.x >> 5);
  const int lane = threadIdx.x & 31;
  if (row >= rows) return;
  const float* x = X + (size_t)row * D_;
  float v[8]; float s = 0.f;
  #pragma unroll
  for (int i = 0; i < 8; ++i) { v[i] = x[lane + i * 32]; s += v[i]; }
  #pragma unroll
  for (int off = 16; off; off >>= 1) s += __shfl_xor(s, off, 32);
  const float mu = s * (1.f / D_);
  float q = 0.f;
  #pragma unroll
  for (int i = 0; i < 8; ++i) { const float dd = v[i] - mu; q += dd * dd; }
  #pragma unroll
  for (int off = 16; off; off >>= 1) q += __shfl_xor(q, off, 32);
  const float inv = rsqrtf(q * (1.f / D_) + 1e-5f);
  #pragma unroll
  for (int i = 0; i < 8; ++i) {
    const int d = lane + i * 32;
    Y[(size_t)row * D_ + d] = (v[i] - mu) * inv * g[d] + be[d];
  }
}

// ---------------- tiny attention (S<=9): one thread per (s,b,h) ----------------
__global__ void k_attn(const float* __restrict__ qkv, float* __restrict__ o, int S)
{
  const int idx = blockIdx.x * blockDim.x + threadIdx.x;
  const int total = B_ * H_ * S;
  if (idx >= total) return;
  const int s = idx / (B_ * H_);
  const int rem = idx - s * (B_ * H_);
  const int b = rem >> 2;
  const int h = rem & 3;
  const int dh = D_ / H_;                 // 64
  const float* q = qkv + ((size_t)(s * B_ + b)) * (3 * D_) + h * dh;
  float lg[16]; float mx = -1e30f;
  for (int t = 0; t < S; ++t) {
    const float* kk = qkv + ((size_t)(t * B_ + b)) * (3 * D_) + D_ + h * dh;
    float dacc = 0.f;
    for (int e = 0; e < dh; ++e) dacc = fmaf(q[e], kk[e], dacc);
    dacc *= 0.125f;                        // 1/sqrt(64)
    lg[t] = dacc; mx = fmaxf(mx, dacc);
  }
  float sum = 0.f;
  for (int t = 0; t < S; ++t) { lg[t] = __expf(lg[t] - mx); sum += lg[t]; }
  const float inv = 1.f / sum;
  float* op = o + ((size_t)(s * B_ + b)) * D_ + h * dh;
  for (int e = 0; e < dh; ++e) {
    float acc = 0.f;
    for (int t = 0; t < S; ++t)
      acc = fmaf(lg[t], qkv[((size_t)(t * B_ + b)) * (3 * D_) + 2 * D_ + h * dh + e], acc);
    op[e] = acc * inv;
  }
}

// ---------------- LSTM cell elementwise ----------------
__global__ void k_lstm(const float* __restrict__ gih, const float* __restrict__ ghh,
                       float* __restrict__ hx, float* __restrict__ cx)
{
  const int b = blockIdx.x, d = threadIdx.x;
  const float gi = gih[b * 4 * D_ + d]            + ghh[b * 4 * D_ + d];
  const float gf = gih[b * 4 * D_ + D_ + d]       + ghh[b * 4 * D_ + D_ + d];
  const float gg = gih[b * 4 * D_ + 2 * D_ + d]   + ghh[b * 4 * D_ + 2 * D_ + d];
  const float go = gih[b * 4 * D_ + 3 * D_ + d]   + ghh[b * 4 * D_ + 3 * D_ + d];
  const float c = sigf(gf) * cx[b * D_ + d] + sigf(gi) * tanhf(gg);
  const float h = sigf(go) * tanhf(c);
  cx[b * D_ + d] = c;
  hx[b * D_ + d] = h;
}

// ---------------- fused dual-key cosine vs all memory rows (one pass over 134MB) ----
__global__ void k_cos(const float* __restrict__ mem,
                      const float* __restrict__ rp, int rps,
                      const float* __restrict__ wp, int wps,
                      float* __restrict__ cr, float* __restrict__ cw)
{
  __shared__ float kr[D_], kw[D_], red[256];
  const int b = blockIdx.y;
  const int tid = threadIdx.x;
  kr[tid] = rp[(size_t)b * rps + tid];
  kw[tid] = wp[(size_t)b * wps + tid];
  __syncthreads();
  red[tid] = kr[tid] * kr[tid]; __syncthreads();
  for (int off = 128; off; off >>= 1) { if (tid < off) red[tid] += red[tid + off]; __syncthreads(); }
  const float nr = red[0]; __syncthreads();
  red[tid] = kw[tid] * kw[tid]; __syncthreads();
  for (int off = 128; off; off >>= 1) { if (tid < off) red[tid] += red[tid + off]; __syncthreads(); }
  const float nw = red[0]; __syncthreads();
  const float knr = sqrtf(nr) + 1e-12f;
  const float knw = sqrtf(nw) + 1e-12f;

  const int n = blockIdx.x * 256 + tid;
  const float* mrow = mem + ((size_t)b * N_ + n) * D_;
  float dr = 0.f, dw = 0.f, mm = 0.f;
  for (int d = 0; d < D_; d += 4) {
    __builtin_prefetch(mrow + d + 64, 0, 0);
    const float4 m4 = *(const float4*)(mrow + d);
    mm = fmaf(m4.x, m4.x, fmaf(m4.y, m4.y, fmaf(m4.z, m4.z, fmaf(m4.w, m4.w, mm))));
    dr = fmaf(m4.x, kr[d], fmaf(m4.y, kr[d+1], fmaf(m4.z, kr[d+2], fmaf(m4.w, kr[d+3], dr))));
    dw = fmaf(m4.x, kw[d], fmaf(m4.y, kw[d+1], fmaf(m4.z, kw[d+2], fmaf(m4.w, kw[d+3], dw))));
  }
  const float mn = sqrtf(mm) + 1e-12f;
  cr[(size_t)b * N_ + n] = dr / (mn * knr);
  cw[(size_t)b * N_ + n] = dw / (mn * knw);
}

// ---------------- NTM addressing: softmax/gate/shift/sharpen, in-place w update ------
__global__ void k_address(const float* __restrict__ cosm,
                          const float* __restrict__ params, int pstride,
                          float* __restrict__ w)
{
  __shared__ float swg[N_];
  __shared__ float red[256];
  const int b = blockIdx.x;
  const int tid = threadIdx.x;
  const float* p = params + (size_t)b * pstride;
  const float beta = softplusf(p[D_]);
  const float gate = sigf(p[D_ + 1]);
  float s0 = p[D_ + 2], s1 = p[D_ + 3], s2 = p[D_ + 4];
  const float smax = fmaxf(s0, fmaxf(s1, s2));
  float e0 = __expf(s0 - smax), e1 = __expf(s1 - smax), e2 = __expf(s2 - smax);
  const float sinv = 1.f / (e0 + e1 + e2);
  e0 *= sinv; e1 *= sinv; e2 *= sinv;
  const float gp = 1.f + softplusf(p[D_ + 5]);

  float cv[16];
  float lmax = -1e30f;
  #pragma unroll
  for (int i = 0; i < 16; ++i) {
    cv[i] = beta * cosm[(size_t)b * N_ + tid + i * 256];
    lmax = fmaxf(lmax, cv[i]);
  }
  red[tid] = lmax; __syncthreads();
  for (int off = 128; off; off >>= 1) { if (tid < off) red[tid] = fmaxf(red[tid], red[tid + off]); __syncthreads(); }
  const float gmax = red[0]; __syncthreads();
  float lsum = 0.f;
  #pragma unroll
  for (int i = 0; i < 16; ++i) { cv[i] = __expf(cv[i] - gmax); lsum += cv[i]; }
  red[tid] = lsum; __syncthreads();
  for (int off = 128; off; off >>= 1) { if (tid < off) red[tid] += red[tid + off]; __syncthreads(); }
  const float inv = 1.f / red[0]; __syncthreads();

  #pragma unroll
  for (int i = 0; i < 16; ++i) {
    const int n = tid + i * 256;
    const float wc = cv[i] * inv;
    swg[n] = gate * wc + (1.f - gate) * w[(size_t)b * N_ + n];
  }
  __syncthreads();

  float wt[16]; float lsum2 = 0.f;
  #pragma unroll
  for (int i = 0; i < 16; ++i) {
    const int n = tid + i * 256;
    // roll(wg, k-1): shifted[n] = s0*wg[n+1] + s1*wg[n] + s2*wg[n-1]  (circular)
    const float sh = e0 * swg[(n + 1) & (N_ - 1)] + e1 * swg[n] + e2 * swg[(n + N_ - 1) & (N_ - 1)];
    const float v = __powf(sh + 1e-12f, gp);
    wt[i] = v; lsum2 += v;
  }
  red[tid] = lsum2; __syncthreads();
  for (int off = 128; off; off >>= 1) { if (tid < off) red[tid] += red[tid + off]; __syncthreads(); }
  const float inv2 = 1.f / (red[0] + 1e-12f);
  #pragma unroll
  for (int i = 0; i < 16; ++i) w[(size_t)b * N_ + tid + i * 256] = wt[i] * inv2;
}

// ---------------- memory erase/add update (one read + one write pass) ----------------
__global__ void k_mem_update(float* __restrict__ mem,
                             const float* __restrict__ ww,
                             const float* __restrict__ wp, int wps)
{
  const int b = blockIdx.y;
  const int nl = threadIdx.x >> 6;           // 4 rows / block
  const int d  = (threadIdx.x & 63) * 4;     // float4 per thread
  const int n  = blockIdx.x * 4 + nl;
  const float wt = ww[(size_t)b * N_ + n];
  const float* p = wp + (size_t)b * wps;
  float4 e, a;
  e.x = sigf(p[262 + d]);     e.y = sigf(p[262 + d + 1]);
  e.z = sigf(p[262 + d + 2]); e.w = sigf(p[262 + d + 3]);
  a.x = tanhf(p[518 + d]);     a.y = tanhf(p[518 + d + 1]);
  a.z = tanhf(p[518 + d + 2]); a.w = tanhf(p[518 + d + 3]);
  float4* mp = (float4*)(mem + ((size_t)b * N_ + n) * D_ + d);
  float4 m = *mp;
  m.x = m.x * (1.f - wt * e.x) + wt * a.x;
  m.y = m.y * (1.f - wt * e.y) + wt * a.y;
  m.z = m.z * (1.f - wt * e.z) + wt * a.z;
  m.w = m.w * (1.f - wt * e.w) + wt * a.w;
  *mp = m;
}

// ==================================================================================
extern "C" void kernel_launch(void* const* d_in, const int* in_sizes, int n_in,
                              void* d_out, int out_size, void* d_ws, size_t ws_size,
                              hipStream_t stream) {
  (void)in_sizes; (void)n_in; (void)out_size; (void)ws_size;

  // ---- inputs (setup_inputs order) ----
  const int*   tokens      = (const int*)d_in[0];
  const float* emb_W       = (const float*)d_in[1];
  const float* read_gate_W = (const float*)d_in[2];
  const float* read_gate_b = (const float*)d_in[3];
  const float* film_W      = (const float*)d_in[4];
  const float* film_b      = (const float*)d_in[5];
  const float* t_Wqkv      = (const float*)d_in[6];
  const float* t_bqkv      = (const float*)d_in[7];
  const float* t_Wo        = (const float*)d_in[8];
  const float* t_bo        = (const float*)d_in[9];
  const float* t_ln1_g     = (const float*)d_in[10];
  const float* t_ln1_b     = (const float*)d_in[11];
  const float* t_ln2_g     = (const float*)d_in[12];
  const float* t_ln2_b     = (const float*)d_in[13];
  const float* t_W1        = (const float*)d_in[14];
  const float* t_b1        = (const float*)d_in[15];
  const float* t_W2        = (const float*)d_in[16];
  const float* t_b2        = (const float*)d_in[17];
  const float* out_ln_g    = (const float*)d_in[18];
  const float* out_ln_b    = (const float*)d_in[19];
  const float* lstm_Wih    = (const float*)d_in[20];
  const float* lstm_Whh    = (const float*)d_in[21];
  const float* lstm_bih    = (const float*)d_in[22];
  const float* lstm_bhh    = (const float*)d_in[23];
  const float* head_W      = (const float*)d_in[24];
  const float* head_b      = (const float*)d_in[25];
  const float* read_W      = (const float*)d_in[26];
  const float* read_b      = (const float*)d_in[27];
  const float* write_W     = (const float*)d_in[28];
  const float* write_b     = (const float*)d_in[29];
  float* out = (float*)d_out;

  // ---- workspace bump allocator ----
  char* wsp = (char*)d_ws;
  auto alloc = [&](size_t bytes) -> char* {
    char* p = wsp; wsp += (bytes + 255) & ~size_t(255); return p;
  };
  float* memory  = (float*)alloc((size_t)B_ * N_ * D_ * 4);
  float* hist    = (float*)alloc((size_t)WIN_ * B_ * D_ * 4);
  float* h_buf   = (float*)alloc((size_t)WIN_ * B_ * D_ * 4);
  float* ln_buf  = (float*)alloc((size_t)WIN_ * B_ * D_ * 4);
  float* qkv_buf = (float*)alloc((size_t)WIN_ * B_ * 3 * D_ * 4);
  float* ff_buf  = (float*)alloc((size_t)WIN_ * B_ * 4 * D_ * 4);
  float* o_buf   = (float*)alloc((size_t)WIN_ * B_ * D_ * 4);
  float* pr_buf  = (float*)alloc((size_t)WIN_ * B_ * D_ * 4);
  float* read_w  = (float*)alloc((size_t)B_ * N_ * 4);
  float* write_w = (float*)alloc((size_t)B_ * N_ * 4);
  float* hx      = (float*)alloc((size_t)2 * B_ * D_ * 4);
  float* cx      = (float*)alloc((size_t)2 * B_ * D_ * 4);
  float* fp      = (float*)alloc((size_t)B_ * 2 * D_ * 4);
  float* rgbuf   = (float*)alloc((size_t)B_ * D_ * 4);
  float* r_vec   = (float*)alloc((size_t)B_ * D_ * 4);
  float* c_out   = (float*)alloc((size_t)B_ * D_ * 4);
  float* cfin    = (float*)alloc((size_t)B_ * D_ * 4);
  float* g_ih    = (float*)alloc((size_t)B_ * 4 * D_ * 4);
  float* g_hh    = (float*)alloc((size_t)B_ * 4 * D_ * 4);
  float* rp      = (float*)alloc((size_t)B_ * 262 * 4);
  float* wp      = (float*)alloc((size_t)B_ * 774 * 4);
  float* cos_r   = (float*)alloc((size_t)B_ * N_ * 4);
  float* cos_w   = (float*)alloc((size_t)B_ * N_ * 4);
  unsigned short* film_bf  = (unsigned short*)alloc((size_t)2 * D_ * D_ * 2);
  unsigned short* rgate_bf = (unsigned short*)alloc((size_t)D_ * D_ * 2);
  unsigned short* qkv_bf   = (unsigned short*)alloc((size_t)L_ * 3 * D_ * D_ * 2);
  unsigned short* wo_bf    = (unsigned short*)alloc((size_t)L_ * D_ * D_ * 2);
  unsigned short* w1_bf    = (unsigned short*)alloc((size_t)L_ * 4 * D_ * D_ * 2);
  unsigned short* w2_bf    = (unsigned short*)alloc((size_t)L_ * D_ * 4 * D_ * 2);
  unsigned short* wih_bf   = (unsigned short*)alloc((size_t)2 * 4 * D_ * D_ * 2);
  unsigned short* whh_bf   = (unsigned short*)alloc((size_t)2 * 4 * D_ * D_ * 2);
  unsigned short* head_bf  = (unsigned short*)alloc((size_t)V_ * D_ * 2);
  unsigned short* readW_bf = (unsigned short*)alloc((size_t)262 * D_ * 2);
  unsigned short* writW_bf = (unsigned short*)alloc((size_t)774 * D_ * 2);

  auto cvt = [&](const float* s, unsigned short* d, size_t n) {
    int g = (int)((n + 255) / 256); if (g > 16384) g = 16384;
    k_cvt_bf16<<<g, 256, 0, stream>>>(s, d, n);
  };
  auto gemm = [&](const float* A, const unsigned short* W, const float* bias,
                  float* C, int M, int Nn, int K) {
    dim3 grid(M / 32, (Nn + 255) / 256);          // M % 32 == 0 at every call site
    if (K == 256)
      k_gemm_bf16<256><<<grid, 256, 0, stream>>>(A, W, bias, C, M, Nn);
    else
      k_gemm_bf16<1024><<<grid, 256, 0, stream>>>(A, W, bias, C, M, Nn);
  };

  // ---- one-time per launch: weight conversion to bf16 ----
  cvt(film_W,      film_bf,  (size_t)2 * D_ * D_);
  cvt(read_gate_W, rgate_bf, (size_t)D_ * D_);
  cvt(t_Wqkv,      qkv_bf,   (size_t)L_ * 3 * D_ * D_);
  cvt(t_Wo,        wo_bf,    (size_t)L_ * D_ * D_);
  cvt(t_W1,        w1_bf,    (size_t)L_ * 4 * D_ * D_);
  cvt(t_W2,        w2_bf,    (size_t)L_ * D_ * 4 * D_);
  cvt(lstm_Wih,    wih_bf,   (size_t)2 * 4 * D_ * D_);
  cvt(lstm_Whh,    whh_bf,   (size_t)2 * 4 * D_ * D_);
  cvt(head_W,      head_bf,  (size_t)V_ * D_);
  cvt(read_W,      readW_bf, (size_t)262 * D_);
  cvt(write_W,     writW_bf, (size_t)774 * D_);

  // ---- state init (deterministic every call) ----
  k_zero<<<8192, 256, 0, stream>>>(memory, (size_t)B_ * N_ * D_);
  k_zero<<<512, 256, 0, stream>>>(hist, (size_t)WIN_ * B_ * D_);
  k_zero<<<64, 256, 0, stream>>>(hx, (size_t)2 * B_ * D_);
  k_zero<<<64, 256, 0, stream>>>(cx, (size_t)2 * B_ * D_);
  k_initw<<<512, 256, 0, stream>>>(read_w, (size_t)B_ * N_);
  k_initw<<<512, 256, 0, stream>>>(write_w, (size_t)B_ * N_);

  const float* record = hx + B_ * D_;   // hx[-1] == layer-1 hidden state

  for (int t = 0; t < T_; ++t) {
    const int S = t + 2;
    const int R = S * B_;

    // --- memory read + controller input ---
    k_read_r<<<B_, 256, 0, stream>>>(memory, read_w, r_vec);
    gemm(record, film_bf,  film_b,      fp,    B_, 2 * D_, D_);
    gemm(record, rgate_bf, read_gate_b, rgbuf, B_, D_,     D_);
    k_xin<<<B_, D_, 0, stream>>>(tokens, emb_W, fp, rgbuf, r_vec, hist, t);
    k_build_h<<<R, D_, 0, stream>>>(hist, h_buf);

    // --- transformer ---
    for (int l = 0; l < L_; ++l) {
      k_ln<<<(R + 7) / 8, 256, 0, stream>>>(h_buf, ln_buf, t_ln1_g + l * D_, t_ln1_b + l * D_, R);
      gemm(ln_buf, qkv_bf + (size_t)l * 3 * D_ * D_, t_bqkv + l * 3 * D_, qkv_buf, R, 3 * D_, D_);
      k_attn<<<(B_ * H_ * S + 127) / 128, 128, 0, stream>>>(qkv_buf, o_buf, S);
      gemm(o_buf, wo_bf + (size_t)l * D_ * D_, t_bo + l * D_, pr_buf, R, D_, D_);
      k_add<<<(R * D_ + 255) / 256, 256, 0, stream>>>(h_buf, pr_buf, (size_t)R * D_);
      k_ln<<<(R + 7) / 8, 256, 0, stream>>>(h_buf, ln_buf, t_ln2_g + l * D_, t_ln2_b + l * D_, R);
      gemm(ln_buf, w1_bf + (size_t)l * 4 * D_ * D_, t_b1 + l * 4 * D_, ff_buf, R, 4 * D_, D_);
      k_relu<<<(R * 4 * D_ + 255) / 256, 256, 0, stream>>>(ff_buf, (size_t)R * 4 * D_);
      gemm(ff_buf, w2_bf + (size_t)l * D_ * 4 * D_, t_b2 + l * D_, pr_buf, R, D_, 4 * D_);
      k_add<<<(R * D_ + 255) / 256, 256, 0, stream>>>(h_buf, pr_buf, (size_t)R * D_);
    }
    k_ln<<<(B_ + 7) / 8, 256, 0, stream>>>(h_buf + (size_t)(S - 1) * B_ * D_, c_out,
                                           out_ln_g, out_ln_b, B_);

    // --- 2-layer LSTM ---
    const float* xl = c_out;
    for (int l = 0; l < 2; ++l) {
      gemm(xl,               wih_bf + (size_t)l * 4 * D_ * D_, lstm_bih + l * 4 * D_, g_ih, B_, 4 * D_, D_);
      gemm(hx + l * B_ * D_, whh_bf + (size_t)l * 4 * D_ * D_, lstm_bhh + l * 4 * D_, g_hh, B_, 4 * D_, D_);
      k_lstm<<<B_, D_, 0, stream>>>(g_ih, g_hh, hx + l * B_ * D_, cx + l * B_ * D_);
      xl = hx + l * B_ * D_;
    }
    k_add2<<<(B_ * D_ + 255) / 256, 256, 0, stream>>>(cfin, c_out, hx + B_ * D_, (size_t)B_ * D_);

    // --- vocab head (bf16 WMMA, head_W resident in L2) ---
    gemm(cfin, head_bf, head_b, out + (size_t)t * B_ * V_, B_, V_, D_);

    // --- addressing params ---
    gemm(cfin, readW_bf, read_b,  rp, B_, 262, D_);
    gemm(cfin, writW_bf, write_b, wp, B_, 774, D_);

    // --- fused cosine (read+write keys, single memory pass) + addressing ---
    k_cos<<<dim3(N_ / 256, B_), 256, 0, stream>>>(memory, rp, 262, wp, 774, cos_r, cos_w);
    k_address<<<B_, 256, 0, stream>>>(cos_r, rp, 262, read_w);
    k_address<<<B_, 256, 0, stream>>>(cos_w, wp, 774, write_w);

    // --- erase/add memory update ---
    k_mem_update<<<dim3(N_ / 4, B_), 256, 0, stream>>>(memory, write_w, wp, 774);
  }
}
